// Attention_9242769621638
// MI455X (gfx1250) — compile-verified, matrix operand order checked by reading, not run on hardware
//
#include <hip/hip_runtime.h>
#include <hip/hip_bf16.h>
#include <cstdint>

// ---------------------------------------------------------------------------
// Attention (B=4, N=2048, DIM=1024, H=16, Dh=64) on MI455X (gfx1250, wave32).
// All matmuls on v_wmma_f32_16x16x32_bf16; flash K/V tiles streamed by the
// Tensor Data Mover (tensor_load_to_lds + s_wait_tensorcnt), double-buffered.
// Workspace (bf16): Q[b,h,n,d] | K[b,h,n,d] | Vt[b,h,d,n] | O[b,n,h*d] = 64 MiB
// ---------------------------------------------------------------------------

#define BQ   4
#define NQ   2048
#define DIMQ 1024
#define HQ   16
#define DHQ  64
#define QK_SCALE 0.03125f  // DIM^-0.5 = 1/32 (module scales by dim, not dim_head)

typedef __attribute__((ext_vector_type(16))) __bf16 v16bf;
typedef __attribute__((ext_vector_type(8)))  __bf16 v8bf;
typedef __attribute__((ext_vector_type(4)))  __bf16 v4bf;
typedef __attribute__((ext_vector_type(8)))  float  v8f;
typedef __attribute__((ext_vector_type(4)))  float  v4f;
typedef __attribute__((ext_vector_type(4)))  unsigned int v4u;
typedef __attribute__((ext_vector_type(8)))  int    v8i;
typedef __attribute__((ext_vector_type(4)))  int    v4i;

#if __has_builtin(__builtin_amdgcn_tensor_load_to_lds)
#define USE_TDM 1
#else
#define USE_TDM 0
#endif

__device__ __forceinline__ v8f vzero8() {
    v8f z;
#pragma unroll
    for (int i = 0; i < 8; ++i) z[i] = 0.0f;
    return z;
}

__device__ __forceinline__ v8f bfmma(v16bf a, v16bf b, v8f c) {
    // (neg_a, A, neg_b, B, c_mod, C, reuse_a, reuse_b)
    return __builtin_amdgcn_wmma_f32_16x16x32_bf16(false, a, false, b,
                                                   (short)0, c, false, false);
}

// 16x32 bf16 fragment (A layout; also valid as B when source is stored
// N-major x K-contiguous). lanes 0-15: rows M=0..15, K {0..7,16..23};
// lanes 16-31: same rows, K {8..15,24..31}. ld multiple of 8, base 16B-aligned.
__device__ __forceinline__ v16bf load_frag(const __bf16* base, int ld, int lane) {
    const int m  = lane & 15;
    const int kh = (lane >> 4) & 1;
    const __bf16* p = base + m * ld + kh * 8;
    v8bf lo = *(const v8bf*)(p);
    v8bf hi = *(const v8bf*)(p + 16);
    v16bf f;
#pragma unroll
    for (int j = 0; j < 8; ++j) { f[j] = lo[j]; f[j + 8] = hi[j]; }
    return f;
}

#if USE_TDM
// Issue one TDM 2D tile load: `rows` x `row_elems` bf16 tile from global
// (row stride `src_stride_elems`) into LDS at byte offset `lds_byte`, padding
// each 128B row chunk with 16B (pad_interval=32 DWORDs, pad_amount=4 DWORDs)
// so 64-elem bf16 rows land with a 72-element (144B) LDS pitch.
__device__ __forceinline__ void tdm_load_2d(const __bf16* gsrc, uint32_t lds_byte,
                                            uint32_t rows, uint32_t row_elems,
                                            uint32_t src_stride_elems) {
    const uint64_t ga = (uint64_t)(uintptr_t)gsrc;
    v4u g0;
    g0[0] = 1u;                                        // count=1, no gather
    g0[1] = lds_byte;                                  // lds_addr
    g0[2] = (uint32_t)(ga & 0xffffffffu);              // global_addr[31:0]
    g0[3] = (uint32_t)((ga >> 32) & 0x01ffffffu)       // global_addr[56:32]
            | (2u << 30);                              // type = 2 ("image")
    v8i g1;
    g1[0] = (int)((1u << 16)                           // data_size = 2 bytes
                | (1u << 20)                           // pad_enable
                | (4u << 22)                           // pad_interval: 32 DWORDs
                | (3u << 25));                         // pad_amount: 4 DWORDs
    g1[1] = (int)(row_elems << 16);                    // tensor_dim0[15:0]
    g1[2] = (int)(rows << 16);                         // tensor_dim1[15:0]
    g1[3] = (int)(row_elems << 16);                    // tile_dim0
    g1[4] = (int)rows;                                 // tile_dim1 (tile_dim2=0)
    g1[5] = (int)src_stride_elems;                     // tensor_dim0_stride
    g1[6] = 0;
    g1[7] = 0;
    v4i z4; z4[0] = z4[1] = z4[2] = z4[3] = 0;
#if defined(__clang_major__) && (__clang_major__ >= 23)
    v8i z8;
#pragma unroll
    for (int i = 0; i < 8; ++i) z8[i] = 0;
    __builtin_amdgcn_tensor_load_to_lds(g0, g1, z4, z4, z8, 0);
#else
    __builtin_amdgcn_tensor_load_to_lds(g0, g1, z4, z4, 0);
#endif
}

__device__ __forceinline__ uint32_t lds_byte_off(const void* p) {
    // gfx1250 flat LDS aperture: low 32 bits of the generic address are the
    // LDS byte offset within the workgroup allocation.
    return (uint32_t)(uintptr_t)p;
}
#endif  // USE_TDM

__device__ __forceinline__ __bf16 cvt_bf16(float x)  { return (__bf16)x; }
__device__ __forceinline__ __bf16 cvt_bf16(__bf16 x) { return x; }

// ---------------------------------------------------------------------------
// 128x128-tile GEMM: C[M,Ncols] = A[M,K] * Bw[K,Ncols]
// EPI==0: A=f32 (x), Bw=f32 (w_qkv); scatter bf16 to Q (x1/32), K, Vt.
// EPI==1: A=bf16 (O), Bw=f32 (w_out); Out = C + bias (f32).
// 8 waves: wave (wm 0..3, wn 0..1) owns a 32x64 sub-tile = 2x4 WMMA frags.
// ---------------------------------------------------------------------------
template <int EPI, typename AT>
__global__ __launch_bounds__(256) void gemm_wmma(
    const AT* __restrict__ A, const float* __restrict__ Bw,
    int M, int K, int Ncols,
    __bf16* __restrict__ Qp, __bf16* __restrict__ Kp, __bf16* __restrict__ Vt,
    const float* __restrict__ bias, float* __restrict__ Out)
{
    constexpr int LDS_LD = 40;                 // 32 + 8 pad (80B rows)
    __shared__ __bf16 As[128 * LDS_LD];        // A tile [m][k]
    __shared__ __bf16 Bs[128 * LDS_LD];        // B tile transposed [n][k]

    const int tid  = threadIdx.x;
    const int lane = tid & 31;
    const int wid  = tid >> 5;
    const int wm   = wid >> 1;
    const int wn   = wid & 1;
    const int m_block = blockIdx.x * 128;
    const int n_block = blockIdx.y * 128;

    v8f c[2][4];
#pragma unroll
    for (int i = 0; i < 2; ++i)
#pragma unroll
        for (int j = 0; j < 4; ++j) c[i][j] = vzero8();

    for (int k0 = 0; k0 < K; k0 += 32) {
        // ---- A tile: vectorized global loads, convert to bf16 ----
        if constexpr (sizeof(AT) == 4) {
#pragma unroll
            for (int i = 0; i < 4; ++i) {
                const int r  = (tid >> 3) + i * 32;
                const int cc = (tid & 7) * 4;
                v4f x = *(const v4f*)&A[(size_t)(m_block + r) * K + k0 + cc];
                v4bf y;
                y[0] = (__bf16)x[0]; y[1] = (__bf16)x[1];
                y[2] = (__bf16)x[2]; y[3] = (__bf16)x[3];
                *(v4bf*)&As[r * LDS_LD + cc] = y;
            }
        } else {
#pragma unroll
            for (int i = 0; i < 2; ++i) {
                const int r  = (tid >> 2) + i * 64;
                const int cc = (tid & 3) * 8;
                *(v8bf*)&As[r * LDS_LD + cc] =
                    *(const v8bf*)&A[(size_t)(m_block + r) * K + k0 + cc];
            }
        }
        // ---- B tile transposed: Bs[n][k] = Bw[k0+k][n_block+n] ----
#pragma unroll
        for (int i = 0; i < 4; ++i) {
            const int kk = (tid >> 5) + i * 8;
            const int nn = (tid & 31) * 4;
            v4f w = *(const v4f*)&Bw[(size_t)(k0 + kk) * Ncols + n_block + nn];
#pragma unroll
            for (int j = 0; j < 4; ++j)
                Bs[(nn + j) * LDS_LD + kk] = (__bf16)w[j];
        }
        if (k0 + 32 < K) {  // global_prefetch_b8 hint for next A tile
            __builtin_prefetch(&A[(size_t)(m_block + (tid >> 1)) * K + k0 + 32], 0, 1);
        }
        __syncthreads();

        v16bf a[2], b[4];
#pragma unroll
        for (int mi = 0; mi < 2; ++mi)
            a[mi] = load_frag(&As[(wm * 32 + mi * 16) * LDS_LD], LDS_LD, lane);
#pragma unroll
        for (int ni = 0; ni < 4; ++ni)
            b[ni] = load_frag(&Bs[(wn * 64 + ni * 16) * LDS_LD], LDS_LD, lane);
#pragma unroll
        for (int mi = 0; mi < 2; ++mi)
#pragma unroll
            for (int ni = 0; ni < 4; ++ni)
                c[mi][ni] = bfmma(a[mi], b[ni], c[mi][ni]);
        __syncthreads();
    }

    // ---- epilogue ----
    const int lh   = lane >> 4;
    const int ncol = lane & 15;
#pragma unroll
    for (int mi = 0; mi < 2; ++mi) {
#pragma unroll
        for (int ni = 0; ni < 4; ++ni) {
#pragma unroll
            for (int r = 0; r < 8; ++r) {
                const int gm = m_block + wm * 32 + mi * 16 + r + 8 * lh;
                const int gn = n_block + wn * 64 + ni * 16 + ncol;
                const float v = c[mi][ni][r];
                if constexpr (EPI == 0) {
                    const int bb    = gm >> 11;
                    const int n     = gm & (NQ - 1);
                    const int which = gn >> 10;         // 0:q 1:k 2:v
                    const int rem   = gn & 1023;
                    const int h     = rem >> 6;
                    const int d     = rem & 63;
                    const size_t bh = (size_t)(bb * HQ + h);
                    if (which == 0)
                        Qp[(bh * NQ + n) * DHQ + d] = (__bf16)(v * QK_SCALE);
                    else if (which == 1)
                        Kp[(bh * NQ + n) * DHQ + d] = (__bf16)v;
                    else
                        Vt[(bh * DHQ + d) * NQ + n] = (__bf16)v;
                } else {
                    Out[(size_t)gm * Ncols + gn] = v + bias[gn];
                }
            }
        }
    }
}

// ---------------------------------------------------------------------------
// Flash attention: one (b,h) per blockIdx.y, 128 query rows per block, one
// 16-row query tile per wave, online softmax over 64-key tiles. K/V tiles
// streamed into double-buffered LDS by the TDM (fallback: cooperative loads).
// ---------------------------------------------------------------------------
__global__ __launch_bounds__(256) void flash_attn(
    const __bf16* __restrict__ Qp, const __bf16* __restrict__ Kp,
    const __bf16* __restrict__ Vt, __bf16* __restrict__ Op)
{
    constexpr int TK = 64;   // keys per tile
    constexpr int KP = 72;   // padded ld (144B rows; matches TDM pad config)
    constexpr int PP = 40;   // P-staging ld
    __shared__ __bf16 Ks[2][TK * KP];     // [key][d]
    __shared__ __bf16 Vs[2][DHQ * KP];    // [d][key]
    __shared__ __bf16 Ps[8][16 * PP];     // per-wave P staging (16 x 32)

    const int tid  = threadIdx.x;
    const int lane = tid & 31;
    const int wid  = tid >> 5;
    const int lh   = lane >> 4;
    const int ncol = lane & 15;
    const int bh   = blockIdx.y;
    const int q0   = blockIdx.x * 128 + wid * 16;

    const __bf16* Qg = Qp + ((size_t)bh * NQ + q0) * DHQ;
    const __bf16* Kg = Kp + (size_t)bh * NQ * DHQ;
    const __bf16* Vg = Vt + (size_t)bh * DHQ * NQ;

    v16bf aq[2];
    aq[0] = load_frag(Qg, DHQ, lane);
    aq[1] = load_frag(Qg + 32, DHQ, lane);

    v8f o[4];
#pragma unroll
    for (int t = 0; t < 4; ++t) o[t] = vzero8();
    float mrow[8], lrow[8];
#pragma unroll
    for (int r = 0; r < 8; ++r) { mrow[r] = -1e30f; lrow[r] = 0.0f; }

#if USE_TDM
    if (wid == 0) {  // prologue: stream tile 0 into buffer 0
        tdm_load_2d(Kg, lds_byte_off(&Ks[0][0]), TK, DHQ, DHQ);
        tdm_load_2d(Vg, lds_byte_off(&Vs[0][0]), DHQ, TK, NQ);
    }
#endif

    for (int kt = 0, t = 0; kt < NQ; kt += TK, ++t) {
        const int cur = t & 1;
#if USE_TDM
        if (wid == 0) {
            if (kt + TK < NQ) {  // stream next tile, then wait for current pair
                tdm_load_2d(Kg + (size_t)(kt + TK) * DHQ,
                            lds_byte_off(&Ks[cur ^ 1][0]), TK, DHQ, DHQ);
                tdm_load_2d(Vg + (kt + TK),
                            lds_byte_off(&Vs[cur ^ 1][0]), DHQ, TK, NQ);
                __builtin_amdgcn_s_wait_tensorcnt(2);
            } else {
                __builtin_amdgcn_s_wait_tensorcnt(0);
            }
        }
        __syncthreads();
#else
        {   // cooperative fill of buffer `cur`: 16 bf16 per thread per tile
            const int r  = tid >> 2;
            const int cb = (tid & 3) * 16;
            const v8bf* sk = (const v8bf*)(Kg + (size_t)(kt + r) * DHQ + cb);
            v8bf k0 = sk[0], k1 = sk[1];
            *(v8bf*)&Ks[cur][r * KP + cb]     = k0;
            *(v8bf*)&Ks[cur][r * KP + cb + 8] = k1;
            const v8bf* sv = (const v8bf*)(Vg + (size_t)r * NQ + kt + cb);
            v8bf v0 = sv[0], v1 = sv[1];
            *(v8bf*)&Vs[cur][r * KP + cb]     = v0;
            *(v8bf*)&Vs[cur][r * KP + cb + 8] = v1;
        }
        __syncthreads();
#endif

        for (int kb = 0; kb < TK; kb += 32) {
            // ---- S = Q K^T for 32 keys: two 16x16 tiles, contraction d=64 ----
            v8f s0 = vzero8(), s1 = vzero8();
#pragma unroll
            for (int kc = 0; kc < 2; ++kc) {
                v16bf bk0 = load_frag(&Ks[cur][(kb + 0)  * KP + kc * 32], KP, lane);
                v16bf bk1 = load_frag(&Ks[cur][(kb + 16) * KP + kc * 32], KP, lane);
                s0 = bfmma(aq[kc], bk0, s0);
                s1 = bfmma(aq[kc], bk1, s1);
            }

            // ---- online softmax (row = VGPR slot r + 8*lane-half) ----
            float pr0[8], pr1[8];
#pragma unroll
            for (int r = 0; r < 8; ++r) {
                float v = fmaxf(s0[r], s1[r]);
                v = fmaxf(v, __shfl_xor(v, 1));
                v = fmaxf(v, __shfl_xor(v, 2));
                v = fmaxf(v, __shfl_xor(v, 4));
                v = fmaxf(v, __shfl_xor(v, 8));
                const float mnew = fmaxf(mrow[r], v);
                const float corr = __expf(mrow[r] - mnew);
                mrow[r] = mnew;
                lrow[r] *= corr;
#pragma unroll
                for (int u = 0; u < 4; ++u) o[u][r] *= corr;
                const float p0 = __expf(s0[r] - mnew);
                const float p1 = __expf(s1[r] - mnew);
                pr0[r] = p0; pr1[r] = p1;
                float ps = p0 + p1;
                ps += __shfl_xor(ps, 1);
                ps += __shfl_xor(ps, 2);
                ps += __shfl_xor(ps, 4);
                ps += __shfl_xor(ps, 8);
                lrow[r] += ps;
            }

            // ---- stage P (16x32) to per-wave LDS, re-read as A-fragment ----
            __bf16* Pw = &Ps[wid][0];
#pragma unroll
            for (int r = 0; r < 8; ++r) {
                Pw[(r + 8 * lh) * PP + ncol]      = (__bf16)pr0[r];
                Pw[(r + 8 * lh) * PP + 16 + ncol] = (__bf16)pr1[r];
            }
            v16bf ap = load_frag(Pw, PP, lane);  // same-wave DS ordering

            // ---- O += P V : 4 output d-tiles, contraction keys=32 ----
#pragma unroll
            for (int u = 0; u < 4; ++u) {
                v16bf bv = load_frag(&Vs[cur][(u * 16) * KP + kb], KP, lane);
                o[u] = bfmma(ap, bv, o[u]);
            }
        }
        __syncthreads();
    }

    // ---- finalize: O /= l, write [b, n, h*64+d] ----
    const int hh = bh & (HQ - 1);
    const int bb = bh >> 4;
#pragma unroll
    for (int r = 0; r < 8; ++r) {
        const float inv = 1.0f / lrow[r];
        const int qrow = q0 + r + 8 * lh;
        __bf16* dst = Op + ((size_t)(bb * NQ + qrow)) * DIMQ + hh * DHQ;
#pragma unroll
        for (int u = 0; u < 4; ++u)
            dst[u * 16 + ncol] = (__bf16)(o[u][r] * inv);
    }
}

// ---------------------------------------------------------------------------
extern "C" void kernel_launch(void* const* d_in, const int* in_sizes, int n_in,
                              void* d_out, int out_size, void* d_ws, size_t ws_size,
                              hipStream_t stream)
{
    const float* x     = (const float*)d_in[0];   // [B, N, DIM]
    const float* w_qkv = (const float*)d_in[1];   // [DIM, 3*H*DH]
    const float* w_out = (const float*)d_in[2];   // [H*DH, DIM]
    const float* b_out = (const float*)d_in[3];   // [DIM]
    float* out = (float*)d_out;                   // [B, N, DIM]

    const size_t REG = (size_t)BQ * HQ * NQ * DHQ;  // 8,388,608 elems
    __bf16* Qp = (__bf16*)d_ws;
    __bf16* Kp = Qp + REG;
    __bf16* Vt = Kp + REG;
    __bf16* Op = Vt + REG;

    const int M = BQ * NQ;  // 8192

    // 1) QKV projection: [8192,1024] x [1024,3072] -> Q,K,Vt (bf16)
    dim3 g1(M / 128, (3 * HQ * DHQ) / 128);
    gemm_wmma<0, float><<<g1, 256, 0, stream>>>(
        x, w_qkv, M, DIMQ, 3 * HQ * DHQ, Qp, Kp, Vt, nullptr, nullptr);

    // 2) Flash attention -> O (bf16, [b, n, h*d])
    dim3 g2(NQ / 128, BQ * HQ);
    flash_attn<<<g2, 256, 0, stream>>>(Qp, Kp, Vt, Op);

    // 3) Output projection + bias: [8192,1024] x [1024,1024] -> f32 out
    dim3 g3(M / 128, DIMQ / 128);
    gemm_wmma<1, __bf16><<<g3, 256, 0, stream>>>(
        Op, w_out, M, DIMQ, DIMQ, nullptr, nullptr, nullptr, b_out, out);
}
